// IndexedConvPCC_75831942578224
// MI455X (gfx1250) — matrix operand order repaired and checked by reading, not compiled, not run on hardware
//
#include <hip/hip_runtime.h>
#include <hip/hip_bf16.h>

#define NPTS   262144
#define PDIM   4
#define CINV   11
#define KF     64
#define NDIMS  3
#define NBINS  256
#define DFFV   256
#define LNEPS  0.001f

typedef __attribute__((ext_vector_type(16))) _Float16 v16h;
typedef __attribute__((ext_vector_type(8)))  float    v8f;
typedef __attribute__((ext_vector_type(4)))  float    v4f;   // trivially constructible 16B chunk

union Frag16 { v16h h; v4f q[2]; unsigned u[8]; };

// Async global->LDS copy of one 16B chunk (gfx1250 GLOBAL_LOAD_ASYNC_TO_LDS_B128,
// tracked by ASYNCcnt). LDS address = low 32 bits of generic pointer (aperture
// truncation rule, ISA 10.2).
__device__ __forceinline__ void async_cp16(void* lds_dst, const void* gsrc) {
  asm volatile("global_load_async_to_lds_b128 %0, %1, off"
               :: "v"((unsigned)(size_t)lds_dst), "v"(gsrc) : "memory");
}
__device__ __forceinline__ void async_wait0() {
  asm volatile("s_wait_asynccnt 0" ::: "memory");
}

// ---------------------------------------------------------------------------
// Prep kernels: fp32 -> f16 conversion, weights stored transposed so that
// B-fragments are contiguous 16B chunks.
// ---------------------------------------------------------------------------
// X (N,4,11) fp32 -> Xin16 (N,4,16) f16, channels 11..15 zero
__global__ void cvt_x_kernel(const float* __restrict__ X, _Float16* __restrict__ Xin) {
  size_t t = (size_t)blockIdx.x * blockDim.x + threadIdx.x;  // over N*4*16
  int c = (int)(t & 15);
  size_t np = t >> 4;
  float v = (c < CINV) ? X[np * CINV + c] : 0.f;
  Xin[t] = (_Float16)v;
}

// cw (3,3,C,64) fp32 -> Wt[d][f][tap*CPAD + c] f16 (KD = padded K dim)
__global__ void cvt_convw_kernel(const float* __restrict__ W, _Float16* __restrict__ Wt,
                                 int C, int CPAD, int KD) {
  int t = blockIdx.x * blockDim.x + threadIdx.x;  // over 3*64*KD
  int kk = t % KD;
  int rest = t / KD;
  int f = rest % 64;
  int d = rest / 64;
  if (d >= NDIMS) return;
  int tap = kk / CPAD, c = kk % CPAD;
  float v = 0.f;
  if (tap < 3 && c < C) v = W[(((size_t)d * 3 + tap) * C + c) * 64 + f];
  Wt[t] = (_Float16)v;
}

// dense W (IN=256, OUT=256) fp32 -> Wt[out][in] f16
__global__ void cvt_dense_kernel(const float* __restrict__ W, _Float16* __restrict__ Wt) {
  int t = blockIdx.x * blockDim.x + threadIdx.x;  // over 65536
  int i = t & 255;
  int o = t >> 8;
  Wt[t] = (_Float16)W[(size_t)i * 256 + o];
}

// Hacc fp32 (N,4,64) + bias -> relu -> H16 f16
__global__ void bias_relu_cvt_kernel(const float* __restrict__ acc, const float* __restrict__ b,
                                     _Float16* __restrict__ out) {
  size_t t = (size_t)blockIdx.x * blockDim.x + threadIdx.x;  // over N*4*64
  float v = acc[t] + b[t & 63];
  out[t] = (_Float16)fmaxf(v, 0.f);
}

// ---------------------------------------------------------------------------
// Indexed conv layer as WMMA GEMM. M = 32 seq positions * P = 128 rows/block.
// Halo rows + weights staged into LDS via async-to-LDS. For dim 0 results are
// plain stores (permutation covers Hacc exactly once); dims 1,2 atomically
// accumulate in a second launch.
// ---------------------------------------------------------------------------
template <int CPAD, int NK>
__global__ __launch_bounds__(256) void conv_wmma_kernel(
    const _Float16* __restrict__ Xin,  // [N][4][CPAD] f16
    const int* __restrict__ I,         // [N][3]
    const _Float16* __restrict__ Wt,   // [3][64][NK*32] f16
    float* __restrict__ Hacc,          // [N][4][64] f32
    int dbase, int accum)
{
  constexpr int KD = NK * 32;
  constexpr int ROW = PDIM * CPAD;     // f16 per gathered sequence row
  constexpr int CPR = ROW / 8;         // 16B chunks per row
  __shared__ __align__(16) _Float16 sX[34 * ROW];
  __shared__ __align__(16) _Float16 sW[64 * KD];

  const int d = blockIdx.z + dbase;
  const int i0 = blockIdx.x * 32;
  const int tid = threadIdx.x;

  // stage weights for this dim (async -> LDS, 16B granularity)
  {
    const _Float16* wsrc = Wt + (size_t)d * 64 * KD;
    for (int t = tid; t < 64 * KD / 8; t += 256)
      async_cp16(sW + t * 8, wsrc + t * 8);
  }
  // stage gathered rows i0-1 .. i0+32 (34 rows); boundary rows zero-filled
  for (int t = tid; t < 34 * CPR; t += 256) {
    int row = t / CPR;
    int col = t - row * CPR;
    int i = i0 - 1 + row;
    _Float16* l = sX + row * ROW + col * 8;
    if (i >= 0 && i < NPTS) {
      int r = I[(size_t)i * NDIMS + d];
      async_cp16(l, Xin + (size_t)r * ROW + col * 8);
    } else {
      v4f z = {};
      *(v4f*)l = z;
    }
  }
  async_wait0();
  __syncthreads();

  const int wid = tid >> 5;        // 8 waves -> 8 row tiles of 16
  const int lane = tid & 31;
  const int lane15 = lane & 15;
  const int hi = lane >> 4;

  // Preload A fragments (16-bit A 16x32 layout: two contiguous 16B chunks
  // at K = ks*32 + hi*8 and +16).
  Frag16 a[NK];
  {
    const int m = wid * 16 + lane15;
    const int il = m >> 2;
    const int p = m & 3;
#pragma unroll
    for (int ks = 0; ks < NK; ++ks) {
#pragma unroll
      for (int half = 0; half < 2; ++half) {
        int kk = ks * 32 + hi * 8 + half * 16;
        int tap = kk / CPAD;
        int c = kk - tap * CPAD;
        if (CPAD != 16 || tap < 3) {
          a[ks].q[half] = *(const v4f*)&sX[(il + tap) * ROW + p * CPAD + c];
        } else {
          v4f z = {};
          a[ks].q[half] = z;
        }
      }
    }
  }

  // Scatter rows for the 8 D-matrix VGPRs.
  int ridx[8], pidx[8];
#pragma unroll
  for (int v = 0; v < 8; ++v) {
    int mm = wid * 16 + v + hi * 8;
    ridx[v] = I[(size_t)(i0 + (mm >> 2)) * NDIMS + d];
    pidx[v] = mm & 3;
  }

#pragma unroll
  for (int nt = 0; nt < 4; ++nt) {
    v8f acc = {};
#pragma unroll
    for (int ks = 0; ks < NK; ++ks) {
      Frag16 b;  // B 32x16 layout: two 16B chunks at K = ks*32 + hi*8 and +16
      const v4f* bp = (const v4f*)&sW[(nt * 16 + lane15) * KD + ks * 32 + hi * 8];
      b.q[0] = bp[0];
      b.q[1] = bp[2];
      acc = __builtin_amdgcn_wmma_f32_16x16x32_f16(false, a[ks].h, false, b.h,
                                                   (short)0, acc, false, false);
    }
    const int f = nt * 16 + lane15;
    if (accum) {
#pragma unroll
      for (int v = 0; v < 8; ++v)
        atomicAdd(&Hacc[(((size_t)ridx[v]) * PDIM + pidx[v]) * KF + f], acc[v]);
    } else {
#pragma unroll
      for (int v = 0; v < 8; ++v)
        Hacc[(((size_t)ridx[v]) * PDIM + pidx[v]) * KF + f] = acc[v];
    }
  }
}

// ---------------------------------------------------------------------------
// Fused residual MLP + LayerNorm + head + softmax*mask.
// 128-row tile per workgroup; 256 KB dynamic LDS (A ping/pong f16 + Y f32).
// ---------------------------------------------------------------------------
__global__ __launch_bounds__(256) void mlp_head_kernel(
    const _Float16* __restrict__ F,     // [N][256] f16
    const _Float16* __restrict__ W0t, const float* __restrict__ b0,
    const _Float16* __restrict__ W1t, const float* __restrict__ b1,
    const _Float16* __restrict__ W2t, const float* __restrict__ b2,
    const float* __restrict__ gamma, const float* __restrict__ beta,
    const _Float16* __restrict__ Hwt, const float* __restrict__ hb,
    const float* __restrict__ Mmask,
    float* __restrict__ out)            // [N][256] f32
{
  extern __shared__ char smem[];
  _Float16* sA0 = (_Float16*)smem;                    // 128*256 f16
  _Float16* sA1 = (_Float16*)(smem + 65536);          // 128*256 f16
  float*    sY  = (float*)(smem + 131072);            // 128*256 f32

  const int tid = threadIdx.x;
  const size_t n0 = (size_t)blockIdx.x * 128;

  __builtin_prefetch(W0t, 0, 1);

  // async-stage F tile (64KB contiguous)
  {
    const _Float16* src = F + n0 * 256;
    for (int t = tid; t < 128 * 32; t += 256)   // 4096 x 16B
      async_cp16(sA0 + t * 8, src + t * 8);
  }
  async_wait0();
  __syncthreads();
  for (int t = tid; t < 128 * 256; t += 256) sY[t] = (float)sA0[t];

  const int wid = tid >> 5;
  const int lane = tid & 31;
  const int lane15 = lane & 15;
  const int hi = lane >> 4;
  const int mrow = wid * 16 + lane15;

  const _Float16* Wts[3] = {W0t, W1t, W2t};
  const float* bias[3] = {b0, b1, b2};
  _Float16* cur = sA0;
  _Float16* nxt = sA1;

  for (int l = 0; l < 3; ++l) {
    __syncthreads();
    const _Float16* Wt = Wts[l];
    const float* bs = bias[l];
    Frag16 a[8];
#pragma unroll
    for (int ks = 0; ks < 8; ++ks) {
      a[ks].q[0] = *(const v4f*)&cur[mrow * 256 + ks * 32 + hi * 8];
      a[ks].q[1] = *(const v4f*)&cur[mrow * 256 + ks * 32 + hi * 8 + 16];
    }
#pragma unroll 1
    for (int nt = 0; nt < 16; ++nt) {
      v8f acc = {};
#pragma unroll
      for (int ks = 0; ks < 8; ++ks) {
        Frag16 b;
        const v4f* bp = (const v4f*)&Wt[(size_t)(nt * 16 + lane15) * 256 + ks * 32 + hi * 8];
        b.q[0] = bp[0];
        b.q[1] = bp[2];
        acc = __builtin_amdgcn_wmma_f32_16x16x32_f16(false, a[ks].h, false, b.h,
                                                     (short)0, acc, false, false);
      }
      int jc = nt * 16 + lane15;
      float bb = bs[jc];
#pragma unroll
      for (int v = 0; v < 8; ++v) {
        int mm = wid * 16 + v + hi * 8;
        float val = fmaxf(acc[v] + bb, 0.f);
        nxt[mm * 256 + jc] = (_Float16)val;
        sY[mm * 256 + jc] += val;
      }
    }
    _Float16* tp = cur; cur = nxt; nxt = tp;
  }
  __syncthreads();

  // LayerNorm over 256 cols -> Z (f16) into cur (each wave owns its 16 rows)
  for (int rr = 0; rr < 16; ++rr) {
    int m = wid * 16 + rr;
    float s1 = 0.f, s2 = 0.f;
#pragma unroll
    for (int jj = 0; jj < 8; ++jj) {
      float v = sY[m * 256 + jj * 32 + lane];
      s1 += v; s2 += v * v;
    }
#pragma unroll
    for (int off = 16; off > 0; off >>= 1) {
      s1 += __shfl_xor(s1, off, 32);
      s2 += __shfl_xor(s2, off, 32);
    }
    float mu = s1 * (1.f / 256.f);
    float var = s2 * (1.f / 256.f) - mu * mu;
    float rstd = rsqrtf(var + LNEPS);
#pragma unroll
    for (int jj = 0; jj < 8; ++jj) {
      int col = jj * 32 + lane;
      float z = (sY[m * 256 + col] - mu) * rstd * gamma[col] + beta[col];
      cur[m * 256 + col] = (_Float16)z;
    }
  }
  __syncthreads();

  // Head GEMM: S = Z @ hw + hb (into sY)
  {
    Frag16 a[8];
#pragma unroll
    for (int ks = 0; ks < 8; ++ks) {
      a[ks].q[0] = *(const v4f*)&cur[mrow * 256 + ks * 32 + hi * 8];
      a[ks].q[1] = *(const v4f*)&cur[mrow * 256 + ks * 32 + hi * 8 + 16];
    }
#pragma unroll 1
    for (int nt = 0; nt < 16; ++nt) {
      v8f acc = {};
#pragma unroll
      for (int ks = 0; ks < 8; ++ks) {
        Frag16 b;
        const v4f* bp = (const v4f*)&Hwt[(size_t)(nt * 16 + lane15) * 256 + ks * 32 + hi * 8];
        b.q[0] = bp[0];
        b.q[1] = bp[2];
        acc = __builtin_amdgcn_wmma_f32_16x16x32_f16(false, a[ks].h, false, b.h,
                                                     (short)0, acc, false, false);
      }
      int jc = nt * 16 + lane15;
      float bb = hb[jc];
#pragma unroll
      for (int v = 0; v < 8; ++v) {
        int mm = wid * 16 + v + hi * 8;
        sY[mm * 256 + jc] = acc[v] + bb;
      }
    }
  }
  __syncthreads();

  // Softmax per row, apply mask, store fp32.
  for (int rr = 0; rr < 16; ++rr) {
    int m = wid * 16 + rr;
    float mx = -3.4e38f;
#pragma unroll
    for (int jj = 0; jj < 8; ++jj) mx = fmaxf(mx, sY[m * 256 + jj * 32 + lane]);
#pragma unroll
    for (int off = 16; off > 0; off >>= 1) mx = fmaxf(mx, __shfl_xor(mx, off, 32));
    float e[8], sum = 0.f;
#pragma unroll
    for (int jj = 0; jj < 8; ++jj) {
      e[jj] = __expf(sY[m * 256 + jj * 32 + lane] - mx);
      sum += e[jj];
    }
#pragma unroll
    for (int off = 16; off > 0; off >>= 1) sum += __shfl_xor(sum, off, 32);
    float inv = 1.f / sum;
    size_t gn = n0 + m;
#pragma unroll
    for (int jj = 0; jj < 8; ++jj) {
      int col = jj * 32 + lane;
      out[gn * 256 + col] = e[jj] * inv * Mmask[col];
    }
  }
}

// ---------------------------------------------------------------------------
extern "C" void kernel_launch(void* const* d_in, const int* in_sizes, int n_in,
                              void* d_out, int out_size, void* d_ws, size_t ws_size,
                              hipStream_t stream) {
  const float* X     = (const float*)d_in[0];
  const int*   I     = (const int*)d_in[1];
  const float* Mmask = (const float*)d_in[2];
  const float* cw0   = (const float*)d_in[3];
  const float* cb0   = (const float*)d_in[4];
  const float* cw1   = (const float*)d_in[5];
  const float* cb1   = (const float*)d_in[6];
  const float* cw2   = (const float*)d_in[7];
  const float* cb2   = (const float*)d_in[8];
  const float* dw0   = (const float*)d_in[9];
  const float* db0   = (const float*)d_in[10];
  const float* dw1   = (const float*)d_in[11];
  const float* db1   = (const float*)d_in[12];
  const float* dw2   = (const float*)d_in[13];
  const float* db2   = (const float*)d_in[14];
  const float* gam   = (const float*)d_in[15];
  const float* bet   = (const float*)d_in[16];
  const float* hw    = (const float*)d_in[17];
  const float* hb    = (const float*)d_in[18];
  float* out = (float*)d_out;

  char* ws = (char*)d_ws;
  size_t off = 0;
  auto alloc = [&](size_t bytes) -> void* {
    off = (off + 255) & ~(size_t)255;
    void* p = ws + off;
    off += bytes;
    return p;
  };

  const size_t NH = (size_t)NPTS * PDIM * KF;
  _Float16* Xin16 = (_Float16*)alloc((size_t)NPTS * PDIM * 16 * 2);
  _Float16* H16a  = (_Float16*)alloc(NH * 2);
  _Float16* H16b  = (_Float16*)alloc(NH * 2);
  float*    Hacc  = (float*)alloc(NH * 4);
  _Float16* Wc0   = (_Float16*)alloc((size_t)NDIMS * 64 * 64 * 2);
  _Float16* Wc1   = (_Float16*)alloc((size_t)NDIMS * 64 * 192 * 2);
  _Float16* Wc2   = (_Float16*)alloc((size_t)NDIMS * 64 * 192 * 2);
  _Float16* Dw0t  = (_Float16*)alloc(65536 * 2);
  _Float16* Dw1t  = (_Float16*)alloc(65536 * 2);
  _Float16* Dw2t  = (_Float16*)alloc(65536 * 2);
  _Float16* Hwt   = (_Float16*)alloc(65536 * 2);

  // ---- weight / input conversion ----
  cvt_x_kernel<<<(NPTS * PDIM * 16) / 256, 256, 0, stream>>>(X, Xin16);
  cvt_convw_kernel<<<(NDIMS * 64 * 64 + 255) / 256, 256, 0, stream>>>(cw0, Wc0, CINV, 16, 64);
  cvt_convw_kernel<<<(NDIMS * 64 * 192 + 255) / 256, 256, 0, stream>>>(cw1, Wc1, 64, 64, 192);
  cvt_convw_kernel<<<(NDIMS * 64 * 192 + 255) / 256, 256, 0, stream>>>(cw2, Wc2, 64, 64, 192);
  cvt_dense_kernel<<<65536 / 256, 256, 0, stream>>>(dw0, Dw0t);
  cvt_dense_kernel<<<65536 / 256, 256, 0, stream>>>(dw1, Dw1t);
  cvt_dense_kernel<<<65536 / 256, 256, 0, stream>>>(dw2, Dw2t);
  cvt_dense_kernel<<<65536 / 256, 256, 0, stream>>>(hw, Hwt);

  dim3 g1(NPTS / 32, 1, 1);   // dim 0: plain stores (full coverage, no zero pass)
  dim3 g2(NPTS / 32, 1, 2);   // dims 1,2: atomic accumulate
  const int nblk = (int)(NH / 256);

  // ---- conv layer 0 ----
  conv_wmma_kernel<16, 2><<<g1, 256, 0, stream>>>(Xin16, I, Wc0, Hacc, 0, 0);
  conv_wmma_kernel<16, 2><<<g2, 256, 0, stream>>>(Xin16, I, Wc0, Hacc, 1, 1);
  bias_relu_cvt_kernel<<<nblk, 256, 0, stream>>>(Hacc, cb0, H16a);

  // ---- conv layer 1 ----
  conv_wmma_kernel<64, 6><<<g1, 256, 0, stream>>>(H16a, I, Wc1, Hacc, 0, 0);
  conv_wmma_kernel<64, 6><<<g2, 256, 0, stream>>>(H16a, I, Wc1, Hacc, 1, 1);
  bias_relu_cvt_kernel<<<nblk, 256, 0, stream>>>(Hacc, cb1, H16b);

  // ---- conv layer 2 ----
  conv_wmma_kernel<64, 6><<<g1, 256, 0, stream>>>(H16b, I, Wc2, Hacc, 0, 0);
  conv_wmma_kernel<64, 6><<<g2, 256, 0, stream>>>(H16b, I, Wc2, Hacc, 1, 1);
  bias_relu_cvt_kernel<<<nblk, 256, 0, stream>>>(Hacc, cb2, H16a);

  // ---- fused MLP + LN + head + softmax (256 KB dynamic LDS per WGP) ----
  (void)hipFuncSetAttribute((const void*)mlp_head_kernel,
                            hipFuncAttributeMaxDynamicSharedMemorySize, 262144);
  mlp_head_kernel<<<NPTS / 128, 256, 262144, stream>>>(
      H16a, Dw0t, db0, Dw1t, db1, Dw2t, db2, gam, bet, Hwt, hb, Mmask, out);

  (void)in_sizes; (void)n_in; (void)out_size; (void)ws_size;
}
